// MultiLatentAttention_54357106098473
// MI455X (gfx1250) — matrix operand order repaired and checked by reading, not compiled, not run on hardware
//
#include <hip/hip_runtime.h>
#include <hip/hip_bf16.h>
#include <math.h>

// ---------------- problem constants (fixed by the reference) ----------------
#define BATCH   2
#define SEQ     2048
#define DMODEL  2048
#define NLAT    512
#define NH      16
#define DH      128
#define DR      32
#define DSPLIT  96
#define BT      (BATCH * SEQ)          // 4096 rows
#define SCALE_ATTN 0.08838834764831845f // 1/sqrt(128)

typedef __bf16 bf16_t;
typedef __attribute__((ext_vector_type(16))) __bf16 v16bf;
typedef __attribute__((ext_vector_type(8)))  __bf16 v8bf;
typedef __attribute__((ext_vector_type(8)))  float  v8f;
typedef __attribute__((ext_vector_type(4)))  int    v4i;

// ---- gfx1250 async global->LDS path (guarded; sync fallback otherwise) ----
#if defined(__has_builtin)
#if __has_builtin(__builtin_amdgcn_global_load_async_to_lds_b128) && \
    __has_builtin(__builtin_amdgcn_s_wait_asynccnt)
#define HAVE_ASYNC_LDS 1
#endif
#endif
#ifndef HAVE_ASYNC_LDS
#define HAVE_ASYNC_LDS 0
#endif

// exact parameter types per hipcc diagnostic: (global v4i*, local v4i*, Ii, Ii)
typedef __attribute__((address_space(1))) v4i* gv4i_p;
typedef __attribute__((address_space(3))) v4i* lv4i_p;

// ---------------- WMMA fragment helpers (CDNA5 bf16 16x16x32) ----------------
// A-matrix 16x32: lane = row (lane&15); lane-group g=lane>>4 holds
// K = {16g..16g+15}  -> two 16B loads at +0 and +8 elements.
__device__ __forceinline__ v16bf frag_a(const bf16_t* p) {
  v8bf c0 = *reinterpret_cast<const v8bf*>(p);
  v8bf c1 = *reinterpret_cast<const v8bf*>(p + 8);
  v16bf r;
#pragma unroll
  for (int i = 0; i < 8; ++i) { r[i] = c0[i]; r[i + 8] = c1[i]; }
  return r;
}
// B-matrix 32x16: lane = column; lane-group g holds 16 contiguous K at 16g.
__device__ __forceinline__ v16bf frag_b(const bf16_t* p) {
  v8bf c0 = *reinterpret_cast<const v8bf*>(p);
  v8bf c1 = *reinterpret_cast<const v8bf*>(p + 8);
  v16bf r;
#pragma unroll
  for (int i = 0; i < 8; ++i) { r[i] = c0[i]; r[i + 8] = c1[i]; }
  return r;
}
// B fragment from the XOR-swizzled LDS tile: row-major [64][32] elems,
// 8-elem piece p stored at physical piece p ^ ((row>>2)&3).
__device__ __forceinline__ v16bf frag_b_lds(const bf16_t* tile, int row, int g) {
  const int s  = (row >> 2) & 3;
  const int p0 = (2 * g) ^ s;          // logical piece 2g
  const int p1 = p0 ^ 1;               // logical piece 2g+1
  v8bf c0 = *reinterpret_cast<const v8bf*>(tile + row * 32 + p0 * 8);
  v8bf c1 = *reinterpret_cast<const v8bf*>(tile + row * 32 + p1 * 8);
  v16bf r;
#pragma unroll
  for (int i = 0; i < 8; ++i) { r[i] = c0[i]; r[i + 8] = c1[i]; }
  return r;
}
__device__ __forceinline__ v8f wmma_bf16(v16bf a, v16bf b, v8f c) {
  return __builtin_amdgcn_wmma_f32_16x16x32_bf16(
      /*neg_a=*/false, a, /*neg_b=*/false, b,
      /*c_mod=*/(short)0, c, /*reuse_a=*/false, /*reuse_b=*/false);
}

// ---------------- f32 -> bf16 convert (optional transpose) ----------------
__global__ void convert_bf16_kernel(const float* __restrict__ src,
                                    bf16_t* __restrict__ dst,
                                    int rows, int cols, int transpose) {
  int i = blockIdx.x * blockDim.x + threadIdx.x;
  int n = rows * cols;
  if (i >= n) return;
  int r = i / cols, c = i - r * cols;
  float v = src[i];
  if (transpose) dst[(size_t)c * rows + r] = (bf16_t)v;
  else           dst[i] = (bf16_t)v;
}

// ---------------- generic WMMA GEMM: C[M,N] = A[M,K] * Bt[N,K]^T + bias ----
// A row-major bf16, Bt row-major bf16 (pre-transposed). Block = 4 waves,
// 64x64 tile; each wave owns 16 rows. The 64x32 B tile (shared by all 4
// waves) is double-buffered in LDS via async global->LDS DMA (ASYNCcnt).
__global__ __launch_bounds__(128)
void gemm_bf16_wmma(const bf16_t* __restrict__ A, const bf16_t* __restrict__ Bt,
                    const float* __restrict__ bias,
                    float* __restrict__ Cf, bf16_t* __restrict__ Cb,
                    int M, int N, int K) {
  __shared__ bf16_t Btile[2][64 * 32];   // 4 KB per buffer
  const int tid  = threadIdx.x;
  const int lane = tid & 31;
  const int wave = tid >> 5;
  const int l16  = lane & 15;
  const int g    = lane >> 4;
  const int n0 = blockIdx.x * 64;
  const int m0 = blockIdx.y * 64 + wave * 16;
  if (m0 >= M || n0 >= N) return;

  // Copy mapping: 256 16-byte chunks per tile; thread tid owns chunks
  // {tid, tid+128}. Chunk c -> row = c>>2, piece = c&3 (8 elems each).
  auto stage = [&](int buf, int k) {
#pragma unroll
    for (int half = 0; half < 2; ++half) {
      int c     = tid + half * 128;
      int row   = c >> 2;
      int piece = c & 3;
      int pp    = piece ^ ((row >> 2) & 3);            // XOR bank swizzle
      const bf16_t* gsrc = Bt + (size_t)(n0 + row) * K + k + piece * 8;
      bf16_t*       ldst = &Btile[buf][row * 32 + pp * 8];
#if HAVE_ASYNC_LDS
      __builtin_amdgcn_global_load_async_to_lds_b128((gv4i_p)gsrc,
                                                     (lv4i_p)ldst, 0, 0);
#else
      *reinterpret_cast<v8bf*>(ldst) = *reinterpret_cast<const v8bf*>(gsrc);
#endif
    }
  };

  const bf16_t* arow = A + (size_t)(m0 + l16) * K + 16 * g;
  v8f acc0 = {}, acc1 = {}, acc2 = {}, acc3 = {};

  const int ntiles = K / 32;
  stage(0, 0);
  for (int i = 0; i < ntiles; ++i) {
    const int k = i * 32;
    if (i + 1 < ntiles) stage((i + 1) & 1, k + 32);
#if HAVE_ASYNC_LDS
    if (i + 1 < ntiles) __builtin_amdgcn_s_wait_asynccnt(2);
    else                __builtin_amdgcn_s_wait_asynccnt(0);
#endif
    __syncthreads();                    // B tile i visible to all 4 waves

    if (k + 128 < K) __builtin_prefetch(arow + k + 128, 0, 0);
    v16bf af = frag_a(arow + k);        // A stays on the direct global path
    const bf16_t* bb = &Btile[i & 1][0];
    acc0 = wmma_bf16(af, frag_b_lds(bb,  0 + l16, g), acc0);
    acc1 = wmma_bf16(af, frag_b_lds(bb, 16 + l16, g), acc1);
    acc2 = wmma_bf16(af, frag_b_lds(bb, 32 + l16, g), acc2);
    acc3 = wmma_bf16(af, frag_b_lds(bb, 48 + l16, g), acc3);

    __syncthreads();                    // reads done before buffer reuse
  }

  v8f accs[4] = {acc0, acc1, acc2, acc3};
#pragma unroll
  for (int t = 0; t < 4; ++t) {
    int n = n0 + t * 16 + l16;
    float bv = bias ? bias[n] : 0.0f;
#pragma unroll
    for (int j = 0; j < 8; ++j) {
      int m = m0 + j + 8 * g;
      float v = accs[t][j] + bv;
      if (Cf) Cf[(size_t)m * N + n] = v;
      if (Cb) Cb[(size_t)m * N + n] = (bf16_t)v;
    }
  }
}

// ---------------- rope + layout pack --------------------------------------
__global__ __launch_bounds__(256)
void pack_rope_kernel(const bf16_t* __restrict__ qbase,
                      const float*  __restrict__ qrot_raw,
                      const bf16_t* __restrict__ kbase,
                      const float*  __restrict__ krot_raw,
                      const bf16_t* __restrict__ vin,
                      bf16_t* __restrict__ Q, bf16_t* __restrict__ Kp,
                      bf16_t* __restrict__ Vt, float* __restrict__ krot_out) {
  int idx = blockIdx.x * blockDim.x + threadIdx.x;
  if (idx >= BT * NH) return;
  int h  = idx & (NH - 1);
  int bt = idx >> 4;                // b*SEQ + t
  int t  = bt & (SEQ - 1);
  int b  = bt >> 11;

  float cs[16], sn[16];
#pragma unroll
  for (int f = 0; f < 16; ++f) {
    // inv_freq = 10000^(-2f/32) = exp(-ln(1e4)*f/16)
    float ang = (float)t * __expf(-0.57564627324485f * (float)f);
    cs[f] = __cosf(ang);
    sn[f] = __sinf(ang);
  }

  size_t qk_off = ((size_t)(b * NH + h) * SEQ + t) * DH;

  // ---- Q: base (96) + roped rot (32)
  {
    const bf16_t* src = qbase + (size_t)bt * (NH * DSPLIT) + h * DSPLIT;
#pragma unroll
    for (int d = 0; d < DSPLIT; ++d) Q[qk_off + d] = src[d];
    const float* xr = qrot_raw + (size_t)bt * (NH * DR) + h * DR;
    float x[DR];
#pragma unroll
    for (int i = 0; i < DR; ++i) x[i] = xr[i];
#pragma unroll
    for (int i = 0; i < DR; ++i) {
      float rot = (i < 16) ? -x[i + 16] : x[i - 16];
      Q[qk_off + DSPLIT + i] = (bf16_t)(x[i] * cs[i & 15] + rot * sn[i & 15]);
    }
  }
  // ---- K: base (96) + roped rot (32); roped values also go to f32 output
  {
    const bf16_t* src = kbase + (size_t)bt * (NH * DSPLIT) + h * DSPLIT;
#pragma unroll
    for (int d = 0; d < DSPLIT; ++d) Kp[qk_off + d] = src[d];
    const float* xr = krot_raw + (size_t)bt * (NH * DR) + h * DR;
    float x[DR];
#pragma unroll
    for (int i = 0; i < DR; ++i) x[i] = xr[i];
#pragma unroll
    for (int i = 0; i < DR; ++i) {
      float rot = (i < 16) ? -x[i + 16] : x[i - 16];
      float v = x[i] * cs[i & 15] + rot * sn[i & 15];
      Kp[qk_off + DSPLIT + i] = (bf16_t)v;
      krot_out[((size_t)bt * NH + h) * DR + i] = v;   // (B,T,H,32)
    }
  }
  // ---- V transpose: Vt[b][h][d][t]
  {
    const bf16_t* src = vin + (size_t)bt * (NH * DH) + h * DH;
    size_t vbase = (size_t)(b * NH + h) * DH * SEQ + t;
#pragma unroll
    for (int d = 0; d < DH; ++d) Vt[vbase + (size_t)d * SEQ] = src[d];
  }
}

// ---------------- flash attention (full softmax, no mask) ------------------
// grid = (SEQ/64, NH, BATCH), block = 128 (4 waves). Each wave: 16 q rows,
// streaming key blocks of 32. 16 wmma per key block per wave.
__global__ __launch_bounds__(128)
void attention_wmma(const bf16_t* __restrict__ Q, const bf16_t* __restrict__ K,
                    const bf16_t* __restrict__ Vt, bf16_t* __restrict__ Obf) {
  __shared__ bf16_t Pl[4][16][32];
  const int lane = threadIdx.x & 31;
  const int wave = threadIdx.x >> 5;
  const int l16  = lane & 15;
  const int g    = lane >> 4;
  const int h = blockIdx.y, b = blockIdx.z;
  const int q0 = blockIdx.x * 64 + wave * 16;

  const bf16_t* Qb = Q  + (size_t)(b * NH + h) * SEQ * DH;
  const bf16_t* Kb = K  + (size_t)(b * NH + h) * SEQ * DH;
  const bf16_t* Vb = Vt + (size_t)(b * NH + h) * DH * SEQ;

  // Q fragments (A operand), 4 chunks of 32 along d
  v16bf qf[4];
  {
    const bf16_t* qrow = Qb + (size_t)(q0 + l16) * DH + 16 * g;
#pragma unroll
    for (int c = 0; c < 4; ++c) qf[c] = frag_a(qrow + c * 32);
  }

  v8f o[8]; float run_max[8], run_sum[8];
#pragma unroll
  for (int dt = 0; dt < 8; ++dt) o[dt] = v8f{};
#pragma unroll
  for (int j = 0; j < 8; ++j) { run_max[j] = -1e30f; run_sum[j] = 0.0f; }

  for (int kb = 0; kb < SEQ; kb += 32) {
    // prefetch next key block rows (streaming read)
    if (kb + 32 < SEQ)
      __builtin_prefetch(Kb + (size_t)(kb + 32 + l16) * DH + 16 * g, 0, 0);

    // scores S[16 q x 32 keys] = two 16-col n-tiles, contraction over d=128
    v8f s0 = {}, s1 = {};
    {
      const bf16_t* kr0 = Kb + (size_t)(kb +      l16) * DH + 16 * g;
      const bf16_t* kr1 = Kb + (size_t)(kb + 16 + l16) * DH + 16 * g;
#pragma unroll
      for (int c = 0; c < 4; ++c) {
        s0 = wmma_bf16(qf[c], frag_b(kr0 + c * 32), s0);
        s1 = wmma_bf16(qf[c], frag_b(kr1 + c * 32), s1);
      }
    }
    // online softmax per row j (row m = j + 8g lives across the 16-lane group)
    float alpha[8];
#pragma unroll
    for (int j = 0; j < 8; ++j) {
      float m = fmaxf(s0[j], s1[j]) * SCALE_ATTN;
#pragma unroll
      for (int d = 1; d < 16; d <<= 1) m = fmaxf(m, __shfl_xor(m, d, 32));
      float nm = fmaxf(run_max[j], m);
      alpha[j] = __expf(run_max[j] - nm);
      run_max[j] = nm;
      float p0 = __expf(s0[j] * SCALE_ATTN - nm);
      float p1 = __expf(s1[j] * SCALE_ATTN - nm);
      s0[j] = p0; s1[j] = p1;
      float rs = p0 + p1;
#pragma unroll
      for (int d = 1; d < 16; d <<= 1) rs += __shfl_xor(rs, d, 32);
      run_sum[j] = run_sum[j] * alpha[j] + rs;
    }
#pragma unroll
    for (int dt = 0; dt < 8; ++dt)
#pragma unroll
      for (int j = 0; j < 8; ++j) o[dt][j] *= alpha[j];

    // D-layout -> A-layout rearrange through LDS (per-wave region, in-order)
#pragma unroll
    for (int j = 0; j < 8; ++j) {
      int row = j + 8 * g;
      Pl[wave][row][l16]      = (bf16_t)s0[j];
      Pl[wave][row][16 + l16] = (bf16_t)s1[j];
    }
    v16bf pf = frag_a(&Pl[wave][l16][16 * g]);

    // out += P[16x32] * V[32 keys x 16 d] for 8 d-tiles (Vt rows contiguous)
#pragma unroll
    for (int dt = 0; dt < 8; ++dt) {
      const bf16_t* vrow = Vb + (size_t)(dt * 16 + l16) * SEQ + kb + 16 * g;
      o[dt] = wmma_bf16(pf, frag_b(vrow), o[dt]);
    }
  }

  // normalize, write bf16 [b*T + q][h*128 + d] for the Wo GEMM
#pragma unroll
  for (int dt = 0; dt < 8; ++dt) {
#pragma unroll
    for (int j = 0; j < 8; ++j) {
      int qrow = q0 + j + 8 * g;
      int d = dt * 16 + l16;
      float v = o[dt][j] / run_sum[j];
      Obf[((size_t)(b * SEQ) + qrow) * (NH * DH) + h * DH + d] = (bf16_t)v;
    }
  }
}

// ---------------- host-side orchestration ----------------------------------
static inline void launch_convert(const float* s, bf16_t* d, int r, int c,
                                  int tr, hipStream_t st) {
  int n = r * c;
  convert_bf16_kernel<<<(n + 255) / 256, 256, 0, st>>>(s, d, r, c, tr);
}
static inline void launch_gemm(const bf16_t* A, const bf16_t* Bt,
                               const float* bias, float* Cf, bf16_t* Cb,
                               int M, int N, int K, hipStream_t st) {
  dim3 grid(N / 64, M / 64);
  gemm_bf16_wmma<<<grid, 128, 0, st>>>(A, Bt, bias, Cf, Cb, M, N, K);
}

extern "C" void kernel_launch(void* const* d_in, const int* in_sizes, int n_in,
                              void* d_out, int out_size, void* d_ws, size_t ws_size,
                              hipStream_t stream) {
  const float* X    = (const float*)d_in[0];
  const float* Wdkv = (const float*)d_in[3];  const float* bdkv = (const float*)d_in[4];
  const float* Wdq  = (const float*)d_in[5];  const float* bdq  = (const float*)d_in[6];
  const float* Wuk  = (const float*)d_in[7];  const float* buk  = (const float*)d_in[8];
  const float* Wuv  = (const float*)d_in[9];  const float* buv  = (const float*)d_in[10];
  const float* Wuq  = (const float*)d_in[11]; const float* buq  = (const float*)d_in[12];
  const float* Wqr  = (const float*)d_in[13]; const float* bqr  = (const float*)d_in[14];
  const float* Wkr  = (const float*)d_in[15]; const float* bkr  = (const float*)d_in[16];
  const float* Wo   = (const float*)d_in[17]; const float* bo   = (const float*)d_in[18];

  float* out_main = (float*)d_out;                       // [B,T,DMODEL]
  float* out_ckv  = out_main + (size_t)BT * DMODEL;      // [B,T,NLAT]
  float* out_krot = out_ckv  + (size_t)BT * NLAT;        // [B,T,NH,DR]

  // workspace carve-out (256B aligned)
  char* w = (char*)d_ws;
  auto alloc = [&](size_t bytes) -> void* {
    void* p = (void*)w; w += (bytes + 255) & ~(size_t)255; return p;
  };
  bf16_t* Xbf     = (bf16_t*)alloc((size_t)BT * DMODEL * 2);
  bf16_t* Wdkv_t  = (bf16_t*)alloc((size_t)DMODEL * NLAT * 2);
  bf16_t* Wdq_t   = (bf16_t*)alloc((size_t)DMODEL * NLAT * 2);
  bf16_t* Wuk_t   = (bf16_t*)alloc((size_t)NLAT * NH * DSPLIT * 2);
  bf16_t* Wuv_t   = (bf16_t*)alloc((size_t)NLAT * NH * DH * 2);
  bf16_t* Wuq_t   = (bf16_t*)alloc((size_t)NLAT * NH * DSPLIT * 2);
  bf16_t* Wqr_t   = (bf16_t*)alloc((size_t)NLAT * NH * DR * 2);
  bf16_t* Wkr_t   = (bf16_t*)alloc((size_t)DMODEL * NH * DR * 2);
  bf16_t* Wo_t    = (bf16_t*)alloc((size_t)NH * DH * DMODEL * 2);
  bf16_t* ckv_bf  = (bf16_t*)alloc((size_t)BT * NLAT * 2);
  bf16_t* cq_bf   = (bf16_t*)alloc((size_t)BT * NLAT * 2);
  bf16_t* kbase_bf= (bf16_t*)alloc((size_t)BT * NH * DSPLIT * 2);
  bf16_t* qbase_bf= (bf16_t*)alloc((size_t)BT * NH * DSPLIT * 2);
  bf16_t* v_bf    = (bf16_t*)alloc((size_t)BT * NH * DH * 2);
  float*  qrot_f  = (float*) alloc((size_t)BT * NH * DR * 4);
  float*  krot_f  = (float*) alloc((size_t)BT * NH * DR * 4);
  bf16_t* Qp      = (bf16_t*)alloc((size_t)BATCH * NH * SEQ * DH * 2);
  bf16_t* Kp      = (bf16_t*)alloc((size_t)BATCH * NH * SEQ * DH * 2);
  bf16_t* Vtp     = (bf16_t*)alloc((size_t)BATCH * NH * DH * SEQ * 2);
  bf16_t* aout_bf = (bf16_t*)alloc((size_t)BT * NH * DH * 2);

  // 1) precision conversion + weight transposition (bf16 [N][K])
  launch_convert(X,    Xbf,    BT,     DMODEL,        0, stream);
  launch_convert(Wdkv, Wdkv_t, DMODEL, NLAT,          1, stream);
  launch_convert(Wdq,  Wdq_t,  DMODEL, NLAT,          1, stream);
  launch_convert(Wuk,  Wuk_t,  NLAT,   NH * DSPLIT,   1, stream);
  launch_convert(Wuv,  Wuv_t,  NLAT,   NH * DH,       1, stream);
  launch_convert(Wuq,  Wuq_t,  NLAT,   NH * DSPLIT,   1, stream);
  launch_convert(Wqr,  Wqr_t,  NLAT,   NH * DR,       1, stream);
  launch_convert(Wkr,  Wkr_t,  DMODEL, NH * DR,       1, stream);
  launch_convert(Wo,   Wo_t,   NH*DH,  DMODEL,        1, stream);

  // 2) projection GEMMs (WMMA bf16, f32 accumulate)
  launch_gemm(Xbf,    Wdkv_t, bdkv, out_ckv, ckv_bf,  BT, NLAT,        DMODEL, stream);
  launch_gemm(Xbf,    Wdq_t,  bdq,  nullptr, cq_bf,   BT, NLAT,        DMODEL, stream);
  launch_gemm(Xbf,    Wkr_t,  bkr,  krot_f,  nullptr, BT, NH * DR,     DMODEL, stream);
  launch_gemm(ckv_bf, Wuk_t,  buk,  nullptr, kbase_bf,BT, NH * DSPLIT, NLAT,   stream);
  launch_gemm(ckv_bf, Wuv_t,  buv,  nullptr, v_bf,    BT, NH * DH,     NLAT,   stream);
  launch_gemm(cq_bf,  Wuq_t,  buq,  nullptr, qbase_bf,BT, NH * DSPLIT, NLAT,   stream);
  launch_gemm(cq_bf,  Wqr_t,  bqr,  qrot_f,  nullptr, BT, NH * DR,     NLAT,   stream);

  // 3) rope + pack Q/K/Vt layouts (+ emit k_rot f32 output)
  pack_rope_kernel<<<(BT * NH) / 256, 256, 0, stream>>>(
      qbase_bf, qrot_f, kbase_bf, krot_f, v_bf, Qp, Kp, Vtp, out_krot);

  // 4) WMMA flash attention
  attention_wmma<<<dim3(SEQ / 64, NH, BATCH), 128, 0, stream>>>(Qp, Kp, Vtp, aout_bf);

  // 5) output projection
  launch_gemm(aout_bf, Wo_t, bo, out_main, nullptr, BT, DMODEL, NH * DH, stream);

  (void)in_sizes; (void)n_in; (void)out_size; (void)ws_size;
}
